// MambaEncoder_1932735283986
// MI455X (gfx1250) — matrix operand order repaired and checked by reading, not compiled, not run on hardware
//
#include <hip/hip_runtime.h>
#include <hip/hip_bf16.h>
#include <math.h>

// ---- model dims (compile-time constants from the reference) ----
#define D_MODEL   512
#define D_STATE   16
#define D_CONV    4
#define NLAYERS   4
#define D_INNER   1024
#define DT_RANK   32
#define BATCHSZ   2
#define SEQLEN    2048
#define M_TOK     (BATCHSZ*SEQLEN)        // 4096 tokens
#define XDBL_W    (DT_RANK + 2*D_STATE)   // 64

typedef __attribute__((ext_vector_type(16))) __bf16        bf16x16;
typedef __attribute__((ext_vector_type(8)))  float         f32x8;
typedef __attribute__((ext_vector_type(4)))  unsigned int  v4u;
typedef __attribute__((ext_vector_type(8)))  int           v8i;
typedef __attribute__((ext_vector_type(4)))  int           v4i;

__device__ __forceinline__ float sigmoidf_(float x) { return 1.0f / (1.0f + __expf(-x)); }

// ---------------- TDM 2D tile load: rows x 32 bf16, row stride = stride_elems ----------------
// D# per CDNA5 ISA: group0 = {flags, lds_addr, global_addr, type=2},
// group1 = {data_size=2B, tensor_dim0=32, tensor_dim1=rows, tile_dim0=32, tile_dim1=rows,
//           tensor_dim0_stride=stride}. 2D => groups 2/3 zero. Tracked by TENSORcnt.
__device__ __forceinline__ void tdm_load_tile(unsigned lds_off, const __bf16* g,
                                              int rows, int stride_elems) {
    unsigned long long ga = (unsigned long long)g;
    v4u g0;
    g0[0] = 1u;                                                   // count=1 (valid), user mode
    g0[1] = lds_off;                                              // lds_addr (bytes)
    g0[2] = (unsigned)ga;                                         // global_addr[31:0]
    g0[3] = (unsigned)((ga >> 32) & 0x01ffffffu) | 0x80000000u;   // global_addr[56:32] | type=2
    v8i g1;
    g1[0] = 0x00010000;                                           // data_size=1 (2 bytes/elem)
    g1[1] = (int)(32u << 16);                                     // tensor_dim0[15:0] = 32
    g1[2] = (int)(((unsigned)rows & 0xffffu) << 16);              // tensor_dim1[15:0] = rows
    g1[3] = (int)(32u << 16);                                     // tensor_dim1 hi=0 | tile_dim0=32
    g1[4] = rows & 0xffff;                                        // tile_dim1=rows, tile_dim2=0
    g1[5] = stride_elems;                                         // tensor_dim0_stride[31:0]
    g1[6] = 0;
    g1[7] = 0;
    v4i z4 = {0, 0, 0, 0};
#if defined(__clang_major__) && (__clang_major__ >= 23)
    v8i z8 = {0, 0, 0, 0, 0, 0, 0, 0};
    __builtin_amdgcn_tensor_load_to_lds(g0, g1, z4, z4, z8, 0);
#else
    __builtin_amdgcn_tensor_load_to_lds(g0, g1, z4, z4, 0);
#endif
}

// ---------------- elementwise fp32 -> bf16 ----------------
__global__ void cvt_bf16_kernel(const float* __restrict__ src, __bf16* __restrict__ dst, int n) {
    int i = blockIdx.x * blockDim.x + threadIdx.x;
    if (i < n) dst[i] = (__bf16)src[i];
}

// ---------------- fp32 [K,N] -> bf16 transposed [N,K] ----------------
__global__ void cvtT_bf16_kernel(const float* __restrict__ src, __bf16* __restrict__ dst,
                                 int K, int N) {
    int i = blockIdx.x * blockDim.x + threadIdx.x;
    if (i >= K * N) return;
    int k = i / N, n = i % N;
    dst[(size_t)n * K + k] = (__bf16)src[i];
}

// ---------------- init residual streams: h_f = x, h_b = flip(x) ----------------
__global__ void init_streams_kernel(const float* __restrict__ x,
                                    float* __restrict__ hf, float* __restrict__ hb) {
    int i = blockIdx.x * blockDim.x + threadIdx.x;
    if (i >= M_TOK * D_MODEL) return;
    int m = i / D_MODEL, j = i % D_MODEL;
    int b = m / SEQLEN, l = m % SEQLEN;
    float v = x[i];
    hf[i] = v;
    hb[(size_t)(b * SEQLEN + (SEQLEN - 1 - l)) * D_MODEL + j] = v;
}

// ---------------- LayerNorm(d_model=512) -> bf16 ----------------
__global__ __launch_bounds__(256) void layernorm_bf16_kernel(
    const float* __restrict__ h, const float* __restrict__ w,
    const float* __restrict__ b, __bf16* __restrict__ out) {
    __shared__ float s_sum[256], s_sq[256];
    int m = blockIdx.x, t = threadIdx.x;
    float v0 = h[(size_t)m * D_MODEL + t];
    float v1 = h[(size_t)m * D_MODEL + t + 256];
    s_sum[t] = v0 + v1;
    s_sq[t]  = v0 * v0 + v1 * v1;
    __syncthreads();
    for (int s = 128; s > 0; s >>= 1) {
        if (t < s) { s_sum[t] += s_sum[t + s]; s_sq[t] += s_sq[t + s]; }
        __syncthreads();
    }
    float mean = s_sum[0] * (1.0f / D_MODEL);
    float var  = s_sq[0] * (1.0f / D_MODEL) - mean * mean;
    float rstd = rsqrtf(var + 1e-5f);
    out[(size_t)m * D_MODEL + t]       = (__bf16)((v0 - mean) * rstd * w[t]       + b[t]);
    out[(size_t)m * D_MODEL + t + 256] = (__bf16)((v1 - mean) * rstd * w[t + 256] + b[t + 256]);
}

// ---------------- WMMA bf16 GEMM with TDM-staged tiles ----------------
// C[M,N] (op)= A[M,K] @ W[K,N], W passed PRE-TRANSPOSED as WT[N,K].
// Block tile 128x64, 8 waves (4x2), 32x32 per wave, BK=32 (one wmma K-step).
// Wave 0 drives two tensor_load_to_lds per K-step, double-buffered against compute.
// EPI: 0 = store, 1 = softplus(acc + bias[n]), 2 = C += acc (residual), 3 = acc + bias[n]
template<int EPI>
__global__ __launch_bounds__(256) void gemm_bf16_kernel(
    const __bf16* __restrict__ A, const __bf16* __restrict__ WT,
    float* __restrict__ C, const float* __restrict__ bias,
    int M, int N, int K) {
    __shared__ __align__(16) __bf16 As[2][128 * 32];   // [m][k] tiles
    __shared__ __align__(16) __bf16 Bs[2][64 * 32];    // [n][k] tiles (WT rows)

    const int t    = threadIdx.x;
    const int wave = t >> 5, lane = t & 31;
    const int wm   = wave & 3, wn = wave >> 2;          // 4x2 wave grid
    const int g    = lane >> 4, ln16 = lane & 15;
    const int m0   = blockIdx.y * 128;
    const int n0   = blockIdx.x * 64;

    // LDS byte offsets (flat LDS address = {aperture_hi, offset_lo} per ISA 10.2)
    const unsigned asOff = (unsigned)(unsigned long long)&As[0][0];
    const unsigned bsOff = (unsigned)(unsigned long long)&Bs[0][0];
    const unsigned ASZ = 128 * 32 * 2, BSZ = 64 * 32 * 2;

    f32x8 acc[2][2];
    #pragma unroll
    for (int a = 0; a < 2; ++a)
        #pragma unroll
        for (int b2 = 0; b2 < 2; ++b2)
            #pragma unroll
            for (int e = 0; e < 8; ++e) acc[a][b2][e] = 0.0f;

    const int nk = K >> 5;
    if (wave == 0) {   // prologue: DMA tile 0 into buffer 0
        tdm_load_tile(asOff, A + (size_t)m0 * K, 128, K);
        tdm_load_tile(bsOff, WT + (size_t)n0 * K, 64, K);
        __builtin_amdgcn_s_wait_tensorcnt(0);
    }
    __syncthreads();

    for (int it = 0; it < nk; ++it) {
        const int p = it & 1;
        if ((it + 1 < nk) && wave == 0) {   // DMA next tile into alternate buffer
            const int k0n = (it + 1) << 5;
            tdm_load_tile(asOff + (1 - p) * ASZ, A + (size_t)m0 * K + k0n, 128, K);
            tdm_load_tile(bsOff + (1 - p) * BSZ, WT + (size_t)n0 * K + k0n, 64, K);
        }

        // A fragment: lane holds row m=lane%16; half i -> k = g*8 + i%8 + (i/8)*16
        bf16x16 af[2], bfv[2];
        #pragma unroll
        for (int mt = 0; mt < 2; ++mt) {
            int base = (wm * 32 + mt * 16 + ln16) * 32 + g * 8;
            ((uint4*)&af[mt])[0] = *(const uint4*)&As[p][base];
            ((uint4*)&af[mt])[1] = *(const uint4*)&As[p][base + 16];
        }
        // B fragment: lane holds col n=lane%16; half i -> k = g*8 + i%8 + (i/8)*16
        #pragma unroll
        for (int nt = 0; nt < 2; ++nt) {
            int base = (wn * 32 + nt * 16 + ln16) * 32 + g * 8;
            ((uint4*)&bfv[nt])[0] = *(const uint4*)&Bs[p][base];
            ((uint4*)&bfv[nt])[1] = *(const uint4*)&Bs[p][base + 16];
        }
        #pragma unroll
        for (int mt = 0; mt < 2; ++mt)
            #pragma unroll
            for (int nt = 0; nt < 2; ++nt)
                acc[mt][nt] = __builtin_amdgcn_wmma_f32_16x16x32_bf16(
                    false, af[mt], false, bfv[nt], (short)0, acc[mt][nt], false, false);

        __syncthreads();                        // everyone done reading buffer p
        if ((it + 1 < nk) && wave == 0)
            __builtin_amdgcn_s_wait_tensorcnt(0);   // next buffer landed
        __syncthreads();
    }

    // C/D layout: VGPR j -> row = g*8 + j, col = lane%16
    #pragma unroll
    for (int mt = 0; mt < 2; ++mt)
        #pragma unroll
        for (int nt = 0; nt < 2; ++nt)
            #pragma unroll
            for (int j = 0; j < 8; ++j) {
                int row = m0 + wm * 32 + mt * 16 + g * 8 + j;
                int col = n0 + wn * 32 + nt * 16 + ln16;
                size_t idx = (size_t)row * N + col;
                float v = acc[mt][nt][j];
                if (EPI == 0) {
                    C[idx] = v;
                } else if (EPI == 1) {
                    float s = v + bias[col];
                    C[idx] = (s > 20.0f) ? s : log1pf(__expf(s));
                } else if (EPI == 2) {
                    C[idx] += v;
                } else {
                    C[idx] = v + bias[col];
                }
            }
}

// ---------------- depthwise causal conv (k=4) + SiLU ----------------
__global__ void conv_silu_kernel(const float* __restrict__ xz, const float* __restrict__ cw,
                                 const float* __restrict__ cb, float* __restrict__ xi,
                                 __bf16* __restrict__ xib) {
    int i = blockIdx.x * blockDim.x + threadIdx.x;
    if (i >= M_TOK * D_INNER) return;
    int m = i / D_INNER, d = i % D_INNER;
    int b = m / SEQLEN, l = m % SEQLEN;
    float acc = cb[d];
    #pragma unroll
    for (int k = 0; k < D_CONV; ++k) {
        int ls = l - (D_CONV - 1) + k;
        if (ls >= 0)
            acc = fmaf(cw[d * D_CONV + k], xz[(size_t)(b * SEQLEN + ls) * (2 * D_INNER) + d], acc);
    }
    float s = acc * sigmoidf_(acc);
    xi[i]  = s;
    xib[i] = (__bf16)s;
}

// ---------------- extract dt-rank slice of xdbl as bf16 ----------------
__global__ void dtin_cvt_kernel(const float* __restrict__ xdbl, __bf16* __restrict__ dtin) {
    int i = blockIdx.x * blockDim.x + threadIdx.x;
    if (i >= M_TOK * DT_RANK) return;
    int m = i / DT_RANK, j = i % DT_RANK;
    dtin[i] = (__bf16)xdbl[(size_t)m * XDBL_W + j];
}

// ---------------- selective scan + D-skip + SiLU(z) gating ----------------
// 4 lanes per (b,d) channel, 4 states each; y reduced via shfl_xor within quad.
__global__ __launch_bounds__(256) void scan_gate_kernel(
    const float* __restrict__ dt, const float* __restrict__ xi,
    const float* __restrict__ xdbl, const float* __restrict__ xz,
    const float* __restrict__ A_log, const float* __restrict__ Dp,
    __bf16* __restrict__ yg) {
    int tid = blockIdx.x * blockDim.x + threadIdx.x;
    if (tid >= BATCHSZ * D_INNER * 4) return;
    int chan = tid >> 2, sub = tid & 3;
    int b = chan / D_INNER, d = chan % D_INNER;
    int n0 = sub * 4;
    float Av[4], h[4] = {0.f, 0.f, 0.f, 0.f};
    #pragma unroll
    for (int j = 0; j < 4; ++j) Av[j] = -__expf(A_log[d * D_STATE + n0 + j]);
    float Dval = Dp[d];
    for (int l = 0; l < SEQLEN; ++l) {
        size_t m = (size_t)b * SEQLEN + l;
        float dtv = dt[m * D_INNER + d];
        float xv  = xi[m * D_INNER + d];
        float y = 0.f;
        #pragma unroll
        for (int j = 0; j < 4; ++j) {
            float Bv = xdbl[m * XDBL_W + DT_RANK + n0 + j];
            float Cv = xdbl[m * XDBL_W + DT_RANK + D_STATE + n0 + j];
            float dA = __expf(dtv * Av[j]);             // trans-pipe, co-executes with VALU
            h[j] = fmaf(dA, h[j], dtv * Bv * xv);
            y    = fmaf(h[j], Cv, y);
        }
        y += __shfl_xor(y, 1, 32);
        y += __shfl_xor(y, 2, 32);
        if (sub == 0) {
            float z   = xz[m * (2 * D_INNER) + D_INNER + d];
            float out = (y + Dval * xv) * (z * sigmoidf_(z));
            yg[m * D_INNER + d] = (__bf16)out;
        }
    }
}

// ---------------- concat(h_f, flip(h_b)) -> bf16 [4096,1024] ----------------
__global__ void concat_cvt_kernel(const float* __restrict__ hf, const float* __restrict__ hb,
                                  __bf16* __restrict__ cat) {
    int i = blockIdx.x * blockDim.x + threadIdx.x;
    if (i >= M_TOK * D_MODEL) return;
    int m = i / D_MODEL, j = i % D_MODEL;
    int b = m / SEQLEN, l = m % SEQLEN;
    cat[(size_t)m * (2 * D_MODEL) + j] = (__bf16)hf[i];
    cat[(size_t)m * (2 * D_MODEL) + D_MODEL + j] =
        (__bf16)hb[(size_t)(b * SEQLEN + (SEQLEN - 1 - l)) * D_MODEL + j];
}

// ---------------- host launch ----------------
extern "C" void kernel_launch(void* const* d_in, const int* in_sizes, int n_in,
                              void* d_out, int out_size, void* d_ws, size_t ws_size,
                              hipStream_t stream) {
    (void)in_sizes; (void)n_in; (void)out_size; (void)ws_size;
    const float* x = (const float*)d_in[0];

    // workspace carving
    char* wsb = (char*)d_ws;
    size_t off = 0;
    auto carve = [&](size_t bytes) -> void* {
        void* p = wsb + off;
        off += (bytes + 255) & ~(size_t)255;
        return p;
    };
    float*  hf    = (float*)carve((size_t)M_TOK * D_MODEL * 4);
    float*  hb    = (float*)carve((size_t)M_TOK * D_MODEL * 4);
    __bf16* hn    = (__bf16*)carve((size_t)M_TOK * D_MODEL * 2);
    float*  xz    = (float*)carve((size_t)M_TOK * 2 * D_INNER * 4);
    float*  xi    = (float*)carve((size_t)M_TOK * D_INNER * 4);
    __bf16* xib   = (__bf16*)carve((size_t)M_TOK * D_INNER * 2);
    float*  xdbl  = (float*)carve((size_t)M_TOK * XDBL_W * 4);
    __bf16* dtin  = (__bf16*)carve((size_t)M_TOK * DT_RANK * 2);
    float*  dtbuf = (float*)carve((size_t)M_TOK * D_INNER * 4);
    __bf16* yg    = (__bf16*)carve((size_t)M_TOK * D_INNER * 2);
    __bf16* cat   = (__bf16*)carve((size_t)M_TOK * 2 * D_MODEL * 2);
    // pre-transposed bf16 weights: [N,K] rows
    __bf16* w_in   = (__bf16*)carve((size_t)2 * NLAYERS * D_MODEL * 2 * D_INNER * 2);
    __bf16* w_xp   = (__bf16*)carve((size_t)2 * NLAYERS * D_INNER * XDBL_W * 2);
    __bf16* w_dt   = (__bf16*)carve((size_t)2 * NLAYERS * DT_RANK * D_INNER * 2);
    __bf16* w_out  = (__bf16*)carve((size_t)2 * NLAYERS * D_INNER * D_MODEL * 2);
    __bf16* w_comb = (__bf16*)carve((size_t)2 * D_MODEL * D_MODEL * 2);

    auto cvtT = [&](const float* src, __bf16* dst, int K, int N) {
        int n = K * N;
        cvtT_bf16_kernel<<<(n + 255) / 256, 256, 0, stream>>>(src, dst, K, N);
    };
    // weight conversion + transpose (GEMM weights only; conv/scan params stay fp32)
    for (int dir = 0; dir < 2; ++dir) {
        int base = 1 + dir * 11;
        for (int layer = 0; layer < NLAYERS; ++layer) {
            int li = dir * NLAYERS + layer;
            cvtT((const float*)d_in[base + 0] + (size_t)layer * D_MODEL * 2 * D_INNER,
                 w_in + (size_t)li * D_MODEL * 2 * D_INNER, D_MODEL, 2 * D_INNER);
            cvtT((const float*)d_in[base + 3] + (size_t)layer * D_INNER * XDBL_W,
                 w_xp + (size_t)li * D_INNER * XDBL_W, D_INNER, XDBL_W);
            cvtT((const float*)d_in[base + 4] + (size_t)layer * DT_RANK * D_INNER,
                 w_dt + (size_t)li * DT_RANK * D_INNER, DT_RANK, D_INNER);
            cvtT((const float*)d_in[base + 8] + (size_t)layer * D_INNER * D_MODEL,
                 w_out + (size_t)li * D_INNER * D_MODEL, D_INNER, D_MODEL);
        }
    }
    cvtT((const float*)d_in[23], w_comb, 2 * D_MODEL, D_MODEL);

    init_streams_kernel<<<(M_TOK * D_MODEL + 255) / 256, 256, 0, stream>>>(x, hf, hb);

    const int EW = 256;
    for (int layer = 0; layer < NLAYERS; ++layer) {
        for (int dir = 0; dir < 2; ++dir) {
            int base = 1 + dir * 11;
            int li   = dir * NLAYERS + layer;
            float* h = dir ? hb : hf;
            const float* lnw  = (const float*)d_in[base + 9]  + layer * D_MODEL;
            const float* lnb  = (const float*)d_in[base + 10] + layer * D_MODEL;
            const float* cw   = (const float*)d_in[base + 1]  + layer * D_INNER * D_CONV;
            const float* cb   = (const float*)d_in[base + 2]  + layer * D_INNER;
            const float* dtb  = (const float*)d_in[base + 5]  + layer * D_INNER;
            const float* Alog = (const float*)d_in[base + 6]  + layer * D_INNER * D_STATE;
            const float* Dpp  = (const float*)d_in[base + 7]  + layer * D_INNER;
            const __bf16* winL  = w_in  + (size_t)li * D_MODEL * 2 * D_INNER;
            const __bf16* wxpL  = w_xp  + (size_t)li * D_INNER * XDBL_W;
            const __bf16* wdtL  = w_dt  + (size_t)li * DT_RANK * D_INNER;
            const __bf16* woutL = w_out + (size_t)li * D_INNER * D_MODEL;

            layernorm_bf16_kernel<<<M_TOK, EW, 0, stream>>>(h, lnw, lnb, hn);
            gemm_bf16_kernel<0><<<dim3(2 * D_INNER / 64, M_TOK / 128), EW, 0, stream>>>(
                hn, winL, xz, nullptr, M_TOK, 2 * D_INNER, D_MODEL);
            conv_silu_kernel<<<(M_TOK * D_INNER + EW - 1) / EW, EW, 0, stream>>>(xz, cw, cb, xi, xib);
            gemm_bf16_kernel<0><<<dim3(XDBL_W / 64, M_TOK / 128), EW, 0, stream>>>(
                xib, wxpL, xdbl, nullptr, M_TOK, XDBL_W, D_INNER);
            dtin_cvt_kernel<<<(M_TOK * DT_RANK + EW - 1) / EW, EW, 0, stream>>>(xdbl, dtin);
            gemm_bf16_kernel<1><<<dim3(D_INNER / 64, M_TOK / 128), EW, 0, stream>>>(
                dtin, wdtL, dtbuf, dtb, M_TOK, D_INNER, DT_RANK);
            scan_gate_kernel<<<(BATCHSZ * D_INNER * 4 + EW - 1) / EW, EW, 0, stream>>>(
                dtbuf, xi, xdbl, xz, Alog, Dpp, yg);
            gemm_bf16_kernel<2><<<dim3(D_MODEL / 64, M_TOK / 128), EW, 0, stream>>>(
                yg, woutL, h, nullptr, M_TOK, D_MODEL, D_INNER);
        }
    }

    concat_cvt_kernel<<<(M_TOK * D_MODEL + EW - 1) / EW, EW, 0, stream>>>(hf, hb, cat);
    gemm_bf16_kernel<3><<<dim3(D_MODEL / 64, M_TOK / 128), EW, 0, stream>>>(
        cat, w_comb, (float*)d_out, (const float*)d_in[24], M_TOK, D_MODEL, 2 * D_MODEL);
}
